// TransformerEncoder_30554397344313
// MI455X (gfx1250) — matrix operand order repaired and checked by reading, not compiled, not run on hardware
//
#include <hip/hip_runtime.h>
#include <hip/hip_bf16.h>

// ---------------------------------------------------------------------------
// Problem constants (match reference)
// ---------------------------------------------------------------------------
#define BDIM 8
#define TDIM 1024
#define EDIM 512
#define HDIM 8
#define HE   (HDIM * EDIM)     // 4096
#define MROW (BDIM * TDIM)     // 8192 token rows
#define NORMF 0.044194173824159216f  // 1/sqrt(512)

// ---------------------------------------------------------------------------
// CDNA5 WMMA types
// ---------------------------------------------------------------------------
typedef __attribute__((ext_vector_type(16))) __bf16       v16bf;
typedef __attribute__((ext_vector_type(8)))  float        v8f;
typedef __attribute__((ext_vector_type(4)))  unsigned int u32x4;
typedef int gvec4i __attribute__((vector_size(16)));       // matches builtin param type

union FragU { u32x4 q[2]; v16bf v; };

// ---- async global->LDS path (gfx1250): use if the builtin exists ----------
#if defined(__has_builtin)
#  if __has_builtin(__builtin_amdgcn_global_load_async_to_lds_b128)
#    define HAS_ASYNC_LDS 1
#  endif
#endif
#ifndef HAS_ASYNC_LDS
#  define HAS_ASYNC_LDS 0
#endif

__device__ __forceinline__ void copy16_g2l(unsigned short* lds, const unsigned short* g) {
#if HAS_ASYNC_LDS
    // param1: int4* in addrspace(1) (non-const); param2: LDS side.
    // Low 32 bits of a generic LDS pointer are the LDS byte offset (ISA aperture rule).
    __builtin_amdgcn_global_load_async_to_lds_b128(
        (__attribute__((address_space(1))) gvec4i*)(unsigned long long)g,
        (__attribute__((address_space(3))) gvec4i*)(unsigned long long)(unsigned int)(unsigned long long)lds,
        0, 0);
#else
    *(u32x4*)lds = *(const u32x4*)g;
#endif
}

__device__ __forceinline__ void wait_async0() {
#if HAS_ASYNC_LDS
#  if __has_builtin(__builtin_amdgcn_s_wait_asynccnt)
    __builtin_amdgcn_s_wait_asynccnt(0);
#  else
    asm volatile("s_wait_asynccnt 0x0" ::: "memory");
#  endif
#endif
}

__device__ __forceinline__ unsigned short f2bf(float f) {
    unsigned int u = __float_as_uint(f);
    u += 0x7FFFu + ((u >> 16) & 1u);          // round-to-nearest-even
    return (unsigned short)(u >> 16);
}

__device__ __forceinline__ float wave_sum(float v) {
    #pragma unroll
    for (int m = 16; m > 0; m >>= 1) v += __shfl_xor(v, m, 32);
    return v;
}
__device__ __forceinline__ float wave_max(float v) {
    #pragma unroll
    for (int m = 16; m > 0; m >>= 1) v = fmaxf(v, __shfl_xor(v, m, 32));
    return v;
}

// ---------------------------------------------------------------------------
// Packing kernels
// ---------------------------------------------------------------------------
__global__ void k_pack_x(const float* __restrict__ x, unsigned short* __restrict__ xbf, long n) {
    long id = (long)blockIdx.x * blockDim.x + threadIdx.x;
    if (id < n) xbf[id] = f2bf(x[id]);
}

// W is K x N row-major fp32; WT is N x K row-major bf16, scaled.
__global__ void k_pack_wT(const float* __restrict__ W, unsigned short* __restrict__ WT,
                          int K, int N, float scale) {
    long id = (long)blockIdx.x * blockDim.x + threadIdx.x;
    if (id >= (long)K * N) return;
    long n = id / K, k = id % K;
    WT[id] = f2bf(W[k * (long)N + n] * scale);
}

__global__ void k_pack_bias2(const float* __restrict__ bq, const float* __restrict__ bk,
                             float* __restrict__ bqk, float scale, int N) {
    int id = blockIdx.x * blockDim.x + threadIdx.x;
    if (id >= 2 * N) return;
    bqk[id] = (id < N ? bq[id] : bk[id - N]) * scale;
}

// ---------------------------------------------------------------------------
// WMMA bf16 GEMM with double-buffered LDS staging:
//   C[M,N] = A[M,K] * B^T[N,K]  (+bias, +relu)
// A row-major (lda), B stored transposed row-major (ldb), batched over z.
// flags: 1 = add bias[n], 2 = relu, 4 = write bf16 (else fp32)
// Block = 128 threads = 4 waves; workgroup tile 128x128, wave tile 64x64.
// Per K-step the workgroup stages A[128x32] and B[128x32] into LDS once
// (async global->LDS when available), halving L2 traffic vs per-wave loads.
// LDS rows padded to 40 halves (20-word stride) -> conflict-free ds_load_b128.
// ---------------------------------------------------------------------------
__global__ __launch_bounds__(128)
void k_gemm_bf16(const unsigned short* __restrict__ A,
                 const unsigned short* __restrict__ Bm,
                 void* __restrict__ Cout,
                 const float* __restrict__ bias,
                 int M, int N, int K,
                 long lda, long ldb, long ldc,
                 long zsA, long zsB, long zsC,
                 int flags) {
    __shared__ __align__(16) unsigned short As[2][128][40];
    __shared__ __align__(16) unsigned short Bs[2][128][40];

    const long tileM0 = (long)blockIdx.y * 128;
    const long tileN0 = (long)blockIdx.x * 128;
    if (tileM0 >= M || tileN0 >= N) return;   // uniform across the block

    const int lane = threadIdx.x & 31;
    const int wave = threadIdx.x >> 5;
    const int wrow = (wave >> 1) * 64;        // wave offset inside LDS tile
    const int wcol = (wave & 1) * 64;
    const long z = blockIdx.z;

    const unsigned short* Ab = A + z * zsA;
    const unsigned short* Bb = Bm + z * zsB;

    v8f acc[4][4];
    #pragma unroll
    for (int i = 0; i < 4; ++i)
        #pragma unroll
        for (int j = 0; j < 4; ++j)
            acc[i][j] = (v8f){0.f, 0.f, 0.f, 0.f, 0.f, 0.f, 0.f, 0.f};

    const int lm     = lane & 15;             // row (A) / col (B) within 16-tile
    const int a_koff = (lane & 16) ? 8  : 0;  // ISA 16-bit A layout halves
    const int b_koff = (lane & 16) ? 8  : 0;  // ISA 16-bit B layout halves

    // cooperative stage of one K-step (128 rows x 32 halves per operand)
    auto stage = [&](int buf, int k0) {
        #pragma unroll
        for (int s = 0; s < 4; ++s) {
            int chunk = s * 128 + (int)threadIdx.x;   // 512 chunks of 16B per operand
            int row = chunk >> 2;
            int ko  = (chunk & 3) * 8;
            copy16_g2l(&As[buf][row][ko], Ab + (tileM0 + row) * lda + k0 + ko);
            copy16_g2l(&Bs[buf][row][ko], Bb + (tileN0 + row) * ldb + k0 + ko);
        }
    };

    stage(0, 0);
    int buf = 0;
    for (int k0 = 0; k0 < K; k0 += 32) {
        wait_async0();          // own async stage for 'buf' complete
        __syncthreads();        // all waves staged 'buf'; all done reading buf^1
        if (k0 + 32 < K) stage(buf ^ 1, k0 + 32);   // overlap next stage with math

        v16bf afrag[4];
        #pragma unroll
        for (int i = 0; i < 4; ++i) {
            const unsigned short* p = &As[buf][wrow + i * 16 + lm][a_koff];
            FragU fu;
            fu.q[0] = *(const u32x4*)(p);        // K chunk [off .. off+7]
            fu.q[1] = *(const u32x4*)(p + 16);   // K chunk [off+16 .. off+23]
            afrag[i] = fu.v;
        }
        #pragma unroll
        for (int j = 0; j < 4; ++j) {
            const unsigned short* p = &Bs[buf][wcol + j * 16 + lm][b_koff];
            FragU fu;
            fu.q[0] = *(const u32x4*)(p);        // K chunk [off .. off+7]
            fu.q[1] = *(const u32x4*)(p + 16);   // K chunk [off+16 .. off+23]
            v16bf bfrag = fu.v;
            #pragma unroll
            for (int i = 0; i < 4; ++i)
                acc[i][j] = __builtin_amdgcn_wmma_f32_16x16x32_bf16(
                    false, afrag[i], false, bfrag, (short)0, acc[i][j], false, false);
        }
        buf ^= 1;
    }

    // Epilogue: C/D layout — lane<16: M=r, N=lane ; lane>=16: M=8+r, N=lane-16
    const int rbase = (lane & 16) ? 8 : 0;
    const long tileM = tileM0 + wrow;
    const long tileN = tileN0 + wcol;
    #pragma unroll
    for (int i = 0; i < 4; ++i) {
        #pragma unroll
        for (int j = 0; j < 4; ++j) {
            #pragma unroll
            for (int r = 0; r < 8; ++r) {
                long gm = tileM + i * 16 + rbase + r;
                long gn = tileN + j * 16 + lm;
                float v = acc[i][j][r];
                if (flags & 1) v += bias[gn];
                if (flags & 2) v = fmaxf(v, 0.f);
                if (flags & 4)
                    ((unsigned short*)Cout)[z * zsC + gm * ldc + gn] = f2bf(v);
                else
                    ((float*)Cout)[z * zsC + gm * ldc + gn] = v;
            }
        }
    }
}

// ---------------------------------------------------------------------------
// Softmax statistics: per row of S (rows x cols), compute max and 1/sum(exp)
// One wave per row.
// ---------------------------------------------------------------------------
__global__ void k_row_max_sum(const float* __restrict__ S, float* __restrict__ mrow,
                              float* __restrict__ invZ, int rows, int cols) {
    int lane = threadIdx.x & 31;
    int wave = threadIdx.x >> 5;
    long row = (long)blockIdx.x * (blockDim.x >> 5) + wave;
    if (row >= rows) return;
    const float* p = S + row * (long)cols;
    float mx = -3.402823466e38f;
    for (int c = lane; c < cols; c += 32) mx = fmaxf(mx, p[c]);
    mx = wave_max(mx);
    float s = 0.f;
    for (int c = lane; c < cols; c += 32) s += __expf(p[c] - mx);
    s = wave_sum(s);
    if (lane == 0) { mrow[row] = mx; invZ[row] = 1.0f / s; }
}

// w[h,k] = sum_q exp(S[h,q,k]-m[h,q]) * invZ[h,q]   (column sums of softmax)
__global__ void k_col_weight(const float* __restrict__ S, const float* __restrict__ mrow,
                             const float* __restrict__ invZ, float* __restrict__ wout,
                             int Hn, int Tn) {
    int id = blockIdx.x * blockDim.x + threadIdx.x;
    if (id >= Hn * Tn) return;
    int h = id / Tn, k = id % Tn;
    const float* Sh = S + (long)h * Tn * Tn;
    const float* mh = mrow + (long)h * Tn;
    const float* zh = invZ + (long)h * Tn;
    float acc = 0.f;
    for (int q = 0; q < Tn; ++q)
        acc += __expf(Sh[(long)q * Tn + k] - mh[q]) * zh[q];
    wout[id] = acc;
}

// y[b,h,c] = sum_k w[b,h,k] * x[b,k,c]
__global__ void k_attn_y(const float* __restrict__ w, const float* __restrict__ x,
                         float* __restrict__ y) {
    int id = blockIdx.x * blockDim.x + threadIdx.x;  // B*H*E threads
    if (id >= BDIM * HDIM * EDIM) return;
    int b = id / (HDIM * EDIM);
    int he = id % (HDIM * EDIM);
    int h = he / EDIM, c = he % EDIM;
    const float* wb = w + ((long)b * HDIM + h) * TDIM;
    const float* xb = x + (long)b * TDIM * EDIM + c;
    float acc = 0.f;
    for (int k = 0; k < TDIM; ++k) acc += wb[k] * xb[(long)k * EDIM];
    y[id] = acc;
}

// out_sum[b,n] = y[b,h,:] . Wv[:,n] + T*bv[n],  h = n/E
__global__ void k_out_sum(const float* __restrict__ y, const float* __restrict__ Wv,
                          const float* __restrict__ bv, float* __restrict__ osum) {
    int id = blockIdx.x * blockDim.x + threadIdx.x;  // B*HE threads
    if (id >= BDIM * HE) return;
    int b = id >> 12, n = id & (HE - 1);
    int h = n >> 9;
    const float* yb = y + ((long)b * HDIM + h) * EDIM;
    float acc = (float)TDIM * bv[n];
    for (int c = 0; c < EDIM; ++c) acc += yb[c] * Wv[(long)c * HE + n];
    osum[id] = acc;
}

// u[b,n] = relu(out_sum[b,:] . Wu[:,n] + bu[n])
__global__ void k_unify(const float* __restrict__ osum, const float* __restrict__ Wu,
                        const float* __restrict__ bu, float* __restrict__ u) {
    int id = blockIdx.x * blockDim.x + threadIdx.x;  // B*E threads
    if (id >= BDIM * EDIM) return;
    int b = id / EDIM, n = id % EDIM;
    const float* ob = osum + (long)b * HE;
    float acc = bu[n];
    for (int c = 0; c < HE; ++c) acc += ob[c] * Wu[(long)c * EDIM + n];
    u[id] = fmaxf(acc, 0.f);
}

// LN1: row = b*T+t ; val = x[row,c] + u[b,c] ; layernorm(g1,b1) -> xln (f32 + bf16)
__global__ void k_ln1(const float* __restrict__ x, const float* __restrict__ u,
                      const float* __restrict__ g, const float* __restrict__ bb,
                      float* __restrict__ xln, unsigned short* __restrict__ xlnbf) {
    int lane = threadIdx.x & 31;
    int wave = threadIdx.x >> 5;
    long row = (long)blockIdx.x * (blockDim.x >> 5) + wave;
    if (row >= MROW) return;
    long b = row / TDIM;
    const float* xr = x + row * EDIM;
    const float* ur = u + b * EDIM;
    float s = 0.f, s2 = 0.f;
    for (int c = lane; c < EDIM; c += 32) {
        float v = xr[c] + ur[c];
        s += v; s2 += v * v;
    }
    s = wave_sum(s); s2 = wave_sum(s2);
    float mu = s * (1.0f / EDIM);
    float var = s2 * (1.0f / EDIM) - mu * mu;
    float rs = rsqrtf(var + 1e-5f);
    for (int c = lane; c < EDIM; c += 32) {
        float v = xr[c] + ur[c];
        float o = (v - mu) * rs * g[c] + bb[c];
        xln[row * EDIM + c] = o;
        xlnbf[row * EDIM + c] = f2bf(o);
    }
}

// LN2: val = xln[row,c] + hbuf[row,c] ; layernorm(g2,b2) -> out
__global__ void k_ln2(const float* __restrict__ xln, const float* __restrict__ hbuf,
                      const float* __restrict__ g, const float* __restrict__ bb,
                      float* __restrict__ out) {
    int lane = threadIdx.x & 31;
    int wave = threadIdx.x >> 5;
    long row = (long)blockIdx.x * (blockDim.x >> 5) + wave;
    if (row >= MROW) return;
    const float* xr = xln + row * EDIM;
    const float* hr = hbuf + row * EDIM;
    float s = 0.f, s2 = 0.f;
    for (int c = lane; c < EDIM; c += 32) {
        float v = xr[c] + hr[c];
        s += v; s2 += v * v;
    }
    s = wave_sum(s); s2 = wave_sum(s2);
    float mu = s * (1.0f / EDIM);
    float var = s2 * (1.0f / EDIM) - mu * mu;
    float rs = rsqrtf(var + 1e-5f);
    for (int c = lane; c < EDIM; c += 32) {
        float v = xr[c] + hr[c];
        out[row * EDIM + c] = (v - mu) * rs * g[c] + bb[c];
    }
}

// ---------------------------------------------------------------------------
// Host launcher
// ---------------------------------------------------------------------------
extern "C" void kernel_launch(void* const* d_in, const int* in_sizes, int n_in,
                              void* d_out, int out_size, void* d_ws, size_t ws_size,
                              hipStream_t stream) {
    (void)in_sizes; (void)n_in; (void)out_size; (void)ws_size;
    const float* x  = (const float*)d_in[0];
    const float* Wq = (const float*)d_in[1];
    const float* bq = (const float*)d_in[2];
    const float* Wk = (const float*)d_in[3];
    const float* bk = (const float*)d_in[4];
    const float* Wv = (const float*)d_in[5];
    const float* bv = (const float*)d_in[6];
    const float* Wu = (const float*)d_in[7];
    const float* bu = (const float*)d_in[8];
    const float* g1 = (const float*)d_in[9];
    const float* b1 = (const float*)d_in[10];
    const float* Wh = (const float*)d_in[11];
    const float* bh = (const float*)d_in[12];
    const float* g2 = (const float*)d_in[13];
    const float* b2 = (const float*)d_in[14];
    float* out = (float*)d_out;

    // ---- workspace layout (bytes, 256-aligned) ----
    char* base = (char*)d_ws;
    size_t o = 0;
    auto take = [&](size_t bytes) { size_t r = o; o = (o + bytes + 255) & ~(size_t)255; return r; };
    unsigned short* xbf   = (unsigned short*)(base + take((size_t)MROW * EDIM * 2));        // 8 MB
    unsigned short* WqkT  = (unsigned short*)(base + take((size_t)2 * HE * EDIM * 2));      // 8 MB
    unsigned short* WhT   = (unsigned short*)(base + take((size_t)EDIM * EDIM * 2));        // 0.5 MB
    float*          bqk   = (float*)(base + take((size_t)2 * HE * 4));                      // 32 KB
    unsigned short* qkbf  = (unsigned short*)(base + take((size_t)MROW * 2 * HE * 2));      // 128 MB
    float*          S     = (float*)(base + take((size_t)HDIM * TDIM * TDIM * 4));          // 32 MB (fits in 192MB L2)
    float*          mrow  = (float*)(base + take((size_t)HDIM * TDIM * 4));
    float*          invZ  = (float*)(base + take((size_t)HDIM * TDIM * 4));
    float*          wbuf  = (float*)(base + take((size_t)BDIM * HDIM * TDIM * 4));
    float*          ybuf  = (float*)(base + take((size_t)BDIM * HDIM * EDIM * 4));
    float*          osum  = (float*)(base + take((size_t)BDIM * HE * 4));
    float*          ubuf  = (float*)(base + take((size_t)BDIM * EDIM * 4));
    float*          xln   = (float*)(base + take((size_t)MROW * EDIM * 4));                 // 16 MB
    unsigned short* xlnbf = (unsigned short*)(base + take((size_t)MROW * EDIM * 2));        // 8 MB
    float*          hbuf  = (float*)(base + take((size_t)MROW * EDIM * 4));                 // 16 MB

    // ---- 1. pack inputs to bf16 (scale 1/sqrt(E) folded into Wq/Wk and biases) ----
    {
        long n = (long)MROW * EDIM;
        k_pack_x<<<dim3((n + 255) / 256), 256, 0, stream>>>(x, xbf, n);
    }
    {
        long n = (long)EDIM * HE;
        k_pack_wT<<<dim3((n + 255) / 256), 256, 0, stream>>>(Wq, WqkT, EDIM, HE, NORMF);
        k_pack_wT<<<dim3((n + 255) / 256), 256, 0, stream>>>(Wk, WqkT + (size_t)HE * EDIM, EDIM, HE, NORMF);
    }
    {
        long n = (long)EDIM * EDIM;
        k_pack_wT<<<dim3((n + 255) / 256), 256, 0, stream>>>(Wh, WhT, EDIM, EDIM, 1.0f);
    }
    k_pack_bias2<<<dim3((2 * HE + 255) / 256), 256, 0, stream>>>(bq, bk, bqk, NORMF, HE);

    // ---- 2. fused Q||K projection: qk[8192, 8192] bf16 = x @ [Wq*n | Wk*n] + [bq*n | bk*n]
    k_gemm_bf16<<<dim3((2 * HE) / 128, MROW / 128, 1), 128, 0, stream>>>(
        xbf, WqkT, qkbf, bqk,
        MROW, 2 * HE, EDIM,
        /*lda*/ EDIM, /*ldb*/ EDIM, /*ldc*/ 2 * HE,
        0, 0, 0, /*flags bias|bf16*/ 1 | 4);

    // ---- 3. per-batch: scores = Q@K^T per head, softmax column-sums -> w[b,h,k]
    for (int b = 0; b < BDIM; ++b) {
        const unsigned short* qb = qkbf + (size_t)b * TDIM * (2 * HE);
        k_gemm_bf16<<<dim3(TDIM / 128, TDIM / 128, HDIM), 128, 0, stream>>>(
            qb, qb + HE, S, nullptr,
            TDIM, TDIM, EDIM,
            /*lda*/ 2 * HE, /*ldb*/ 2 * HE, /*ldc*/ TDIM,
            /*zsA head*/ EDIM, /*zsB head*/ EDIM, /*zsC*/ (long)TDIM * TDIM,
            /*flags*/ 0);
        k_row_max_sum<<<dim3((HDIM * TDIM) / 8), 256, 0, stream>>>(S, mrow, invZ, HDIM * TDIM, TDIM);
        k_col_weight<<<dim3((HDIM * TDIM + 255) / 256), 256, 0, stream>>>(
            S, mrow, invZ, wbuf + (size_t)b * HDIM * TDIM, HDIM, TDIM);
    }

    // ---- 4. y[b,h,:] = sum_k w[b,h,k] x[b,k,:]  (attn·V collapsed via query-sum) ----
    k_attn_y<<<dim3((BDIM * HDIM * EDIM + 255) / 256), 256, 0, stream>>>(wbuf, x, ybuf);

    // ---- 5. out_sum = y @ Wv (per head slice) + T*bv ----
    k_out_sum<<<dim3((BDIM * HE + 255) / 256), 256, 0, stream>>>(ybuf, Wv, bv, osum);

    // ---- 6. u = relu(out_sum @ Wu + bu) ----
    k_unify<<<dim3((BDIM * EDIM + 255) / 256), 256, 0, stream>>>(osum, Wu, bu, ubuf);

    // ---- 7. LN1: xln = layernorm(x + u_broadcast)  (+ bf16 copy) ----
    k_ln1<<<dim3(MROW / 8), 256, 0, stream>>>(x, ubuf, g1, b1, xln, xlnbf);

    // ---- 8. hbuf = relu(xln @ Wh + bh) via WMMA ----
    k_gemm_bf16<<<dim3(EDIM / 128, MROW / 128, 1), 128, 0, stream>>>(
        xlnbf, WhT, hbuf, bh,
        MROW, EDIM, EDIM,
        /*lda*/ EDIM, /*ldb*/ EDIM, /*ldc*/ EDIM,
        0, 0, 0, /*flags bias|relu*/ 1 | 2);

    // ---- 9. LN2 -> out ----
    k_ln2<<<dim3(MROW / 8), 256, 0, stream>>>(xln, hbuf, g2, b2, out);
}
